// VisionAttention_28063316312072
// MI455X (gfx1250) — matrix operand order repaired and checked by reading, not compiled, not run on hardware
//
#include <hip/hip_runtime.h>
#include <hip/hip_bf16.h>

typedef _Float16 v16h __attribute__((ext_vector_type(16)));
typedef float    v8f  __attribute__((ext_vector_type(8)));
typedef unsigned u32x4 __attribute__((ext_vector_type(4)));
typedef int      i32x4 __attribute__((ext_vector_type(4)));
typedef int      i32x8 __attribute__((ext_vector_type(8)));

#define WMMA_F32_F16(a, b, c) \
  __builtin_amdgcn_wmma_f32_16x16x32_f16(false, (a), false, (b), (short)0, (c), false, false)

constexpr int NTOK = 8192;
constexpr int DIM  = 1280;
constexpr int NH   = 16;
constexpr int HD   = 80;    // head dim
constexpr int HDP  = 128;   // padded head dim
constexpr int SEQ  = 2048;  // tokens per batch (b = 4)

// ---------------------------------------------------------------------------
// TDM: issue one descriptor loading a [128 rows x 32 halves] tile (row stride
// K halves) from global f16 into LDS at byte offset ldsOff. D# per ISA 8.3/8.4:
// group0 = {count=1 | lds_addr | global_addr | type=2},
// group1 = {data_size=2B, tensor_dim0=K, tile 32x128, dim0_stride=K}.
// ---------------------------------------------------------------------------
__device__ __forceinline__ void tdm_load_tile(const _Float16* gptr,
                                              unsigned ldsOff, int K) {
  const unsigned long long ga = (unsigned long long)(size_t)gptr;
  u32x4 g0;
  g0.x = 1u;                                               // count=1, user mode
  g0.y = ldsOff;                                           // lds_addr
  g0.z = (unsigned)ga;                                     // global_addr lo
  g0.w = ((unsigned)(ga >> 32) & 0x01FFFFFFu) | (2u << 30);// global_addr hi | type=2
  const unsigned td0 = (unsigned)K;      // tensor_dim0 (elements)
  const unsigned td1 = 1u << 20;         // tensor_dim1 (large; tiles never OOB)
  i32x8 g1;
  g1[0] = (int)(1u << 16);                                  // data_size = 1 (2B)
  g1[1] = (int)((td0 & 0xFFFFu) << 16);                     // tensor_dim0 lo
  g1[2] = (int)((td0 >> 16) | ((td1 & 0xFFFFu) << 16));     // dim0 hi | dim1 lo
  g1[3] = (int)((td1 >> 16) | (32u << 16));                 // dim1 hi | tile_dim0=32
  g1[4] = (int)128u;                                        // tile_dim1=128, tile_dim2=0
  g1[5] = (int)td0;                                         // tensor_dim0_stride lo32 = K
  g1[6] = 0;                                                // stride0 hi | stride1 lo
  g1[7] = 0;                                                // stride1 hi
  i32x4 z4 = {0, 0, 0, 0};
#if __clang_major__ >= 23
  i32x8 z8 = {0, 0, 0, 0, 0, 0, 0, 0};
  __builtin_amdgcn_tensor_load_to_lds(g0, g1, z4, z4, z8, 0);
#else
  __builtin_amdgcn_tensor_load_to_lds(g0, g1, z4, z4, 0);
#endif
}

// ---------------------------------------------------------------------------
// f32 -> f16 conversion (vectorized, n % 4 == 0)
// ---------------------------------------------------------------------------
__global__ __launch_bounds__(256) void f32_to_f16(const float* __restrict__ src,
                                                  _Float16* __restrict__ dst, int n) {
  const int i = (blockIdx.x * 256 + threadIdx.x) * 4;
  if (i < n) {
    const float4 v = *(const float4*)(src + i);
    union { _Float16 h[4]; uint2 u; } t;
    t.h[0] = (_Float16)v.x; t.h[1] = (_Float16)v.y;
    t.h[2] = (_Float16)v.z; t.h[3] = (_Float16)v.w;
    *(uint2*)(dst + i) = t.u;
  }
}

// ---------------------------------------------------------------------------
// GEMM: C[M,N] = A[M,K]f16 * W[N,K]^T f16 + bias, f16 or f32 out.
// 128x128x32 tiles, TDM double-buffered global->LDS, 8 waves x (4x2) WMMA.
// A-fragment uses contiguous k: lane<16 K=0-15, lane>=16
// K=16-31; B-fragment is 16 contiguous halves per half-wave.
// ---------------------------------------------------------------------------
__global__ __launch_bounds__(256) void gemm_bias(const _Float16* __restrict__ A,
                                                 const _Float16* __restrict__ W,
                                                 const float* __restrict__ bias,
                                                 _Float16* __restrict__ C16,
                                                 float* __restrict__ C32,
                                                 int N, int K) {
  __shared__ _Float16 As[2][128][32];
  __shared__ _Float16 Bs[2][128][32];

  const int tid    = threadIdx.x;
  const int lane   = tid & 31;
  const int wid    = tid >> 5;
  const int wm     = wid >> 2;           // 0..1
  const int wn     = wid & 3;            // 0..3
  const int m0     = blockIdx.y * 128;
  const int n0     = blockIdx.x * 128;
  const int laneM  = lane & 15;
  const int laneKb = lane & 16;          // B-fragment offset (halves)
  const int laneKa = laneKb;             // A-fragment first chunk {0,16}
  const unsigned asBase = (unsigned)(size_t)&As[0][0][0];
  const unsigned bsBase = (unsigned)(size_t)&Bs[0][0][0];
  const int NK = K / 32;

  v8f acc[4][2] = {};

  if (wid == 0) {  // prologue: stage 0 into buffer 0
    tdm_load_tile(A + (size_t)m0 * K, asBase, K);
    tdm_load_tile(W + (size_t)n0 * K, bsBase, K);
  }

  for (int ks = 0; ks < NK; ks++) {
    const int buf = ks & 1;
    if (wid == 0) {
      if (ks + 1 < NK) {  // prefetch next stage, then wait for current (in-order)
        const unsigned nb = (unsigned)(buf ^ 1) * (128u * 32u * 2u);
        tdm_load_tile(A + (size_t)m0 * K + (ks + 1) * 32, asBase + nb, K);
        tdm_load_tile(W + (size_t)n0 * K + (ks + 1) * 32, bsBase + nb, K);
        __builtin_amdgcn_s_wait_tensorcnt(2);
      } else {
        __builtin_amdgcn_s_wait_tensorcnt(0);
      }
    }
    __syncthreads();

    union AV { v16h h; uint4 q[2]; };
    v16h af[4], bf[2];
    for (int mt = 0; mt < 4; mt++) {
      const _Float16* rp = &As[buf][wm * 64 + mt * 16 + laneM][0];
      AV a;
      a.q[0] = *(const uint4*)(rp + laneKa);
      a.q[1] = *(const uint4*)(rp + laneKa + 8);
      af[mt] = a.h;
    }
    for (int nt = 0; nt < 2; nt++)
      bf[nt] = *(const v16h*)&Bs[buf][wn * 32 + nt * 16 + laneM][laneKb];
    for (int mt = 0; mt < 4; mt++)
      for (int nt = 0; nt < 2; nt++)
        acc[mt][nt] = WMMA_F32_F16(af[mt], bf[nt], acc[mt][nt]);

    __syncthreads();  // all waves done with buf before it is overwritten
  }

  const int rbase = (lane & 16) >> 1;  // D layout: VGPR r -> row r (+8 high half)
  for (int mt = 0; mt < 4; mt++)
    for (int nt = 0; nt < 2; nt++) {
      const int n  = n0 + wn * 32 + nt * 16 + laneM;
      const float bb = bias[n];
      const int mbase = m0 + wm * 64 + mt * 16 + rbase;
      for (int r = 0; r < 8; r++) {
        const float v = acc[mt][nt][r] + bb;
        const size_t idx = (size_t)(mbase + r) * N + n;
        if (C16) C16[idx] = (_Float16)v;
        else     C32[idx] = v;
      }
    }
}

// ---------------------------------------------------------------------------
// RoPE + repack: qkv f16 [8192][3840] -> Q/K [bh][seq][128] (rope, *80^-1/4),
// V transposed [bh][128][seq]. Head dim padded 80->128 with zeros.
// ---------------------------------------------------------------------------
__global__ __launch_bounds__(256) void rope_pack(const _Float16* __restrict__ qkv,
                                                 const float* __restrict__ rot,
                                                 _Float16* __restrict__ Qh,
                                                 _Float16* __restrict__ Kh,
                                                 _Float16* __restrict__ Vt) {
  const int g   = blockIdx.x * 256 + threadIdx.x;
  const int j   = g & (HDP - 1);
  const int h   = (g >> 7) & (NH - 1);
  const int tok = g >> 11;
  const int b   = tok >> 11;
  const int seq = tok & (SEQ - 1);
  const size_t bh   = (size_t)b * NH + h;
  const size_t qidx = (bh * SEQ + seq) * HDP + j;
  const size_t vidx = (bh * HDP + j) * SEQ + seq;

  if (j < HD) {
    const size_t base = (size_t)tok * (3 * DIM) + h * HD;
    const float qv = (float)qkv[base + j];
    const float kv = (float)qkv[base + DIM + j];
    const float vv = (float)qkv[base + 2 * DIM + j];
    const int   jm = (j < 40) ? j : j - 40;
    const float r  = rot[tok * 40 + jm];
    const float c  = __cosf(r), s = __sinf(r);
    const int   jp = (j < 40) ? j + 40 : j - 40;
    const float qp = (float)qkv[base + jp];
    const float kp = (float)qkv[base + DIM + jp];
    const float sg = (j < 40) ? -1.f : 1.f;
    const float sc = 0.33437015f;  // 80^(-1/4) folded into q and k
    Qh[qidx] = (_Float16)((qv * c + sg * qp * s) * sc);
    Kh[qidx] = (_Float16)((kv * c + sg * kp * s) * sc);
    Vt[vidx] = (_Float16)vv;
  } else {
    Qh[qidx] = (_Float16)0.f;
    Kh[qidx] = (_Float16)0.f;
    Vt[vidx] = (_Float16)0.f;
  }
}

// ---------------------------------------------------------------------------
// Flash attention, transposed: S^T = K*Q^T (softmax axis -> VGPRs + one
// half-wave swap), O^T = V^T*P. One wave per (bh, 16 q rows). f16 out.
// ---------------------------------------------------------------------------
__global__ __launch_bounds__(32) void attn(const _Float16* __restrict__ Qh,
                                           const _Float16* __restrict__ Kh,
                                           const _Float16* __restrict__ Vt,
                                           _Float16* __restrict__ O) {
  const int lane   = threadIdx.x;
  const int qb     = blockIdx.x & 127;
  const int bh     = blockIdx.x >> 7;
  const int q0     = qb * 16;
  const int laneM  = lane & 15;
  const int laneKb = lane & 16;       // B-fragment contiguous offset
  const int laneKa = laneKb;          // A-fragment chunk base {0,16}

  // Q fragments: WMMA *B* operand (contiguous 16 halves per half-wave).
  const _Float16* qbase = Qh + ((size_t)bh * SEQ + q0 + laneM) * HDP + laneKb;
  v16h qf[4];
  for (int ds = 0; ds < 4; ds++) qf[ds] = *(const v16h*)(qbase + ds * 32);

  v8f o[8] = {};
  float mrow = -1e30f, lrow = 0.f;

  const _Float16* kbh = Kh + (size_t)bh * SEQ * HDP;
  const _Float16* vbh = Vt + (size_t)bh * HDP * SEQ;

  union AV { v16h h; uint4 q[2]; };

  for (int kv0 = 0; kv0 < SEQ; kv0 += 32) {
    // S^T tiles [kv 32 x q 16]; K rows are the WMMA A operand (contiguous layout).
    v8f st[2] = {};
    for (int mt = 0; mt < 2; mt++) {
      const _Float16* kp = kbh + (size_t)(kv0 + mt * 16 + laneM) * HDP + laneKa;
      for (int ds = 0; ds < 4; ds++) {
        AV a;
        a.q[0] = *(const uint4*)(kp + ds * 32);
        a.q[1] = *(const uint4*)(kp + ds * 32 + 8);
        st[mt] = WMMA_F32_F16(a.h, qf[ds], st[mt]);
      }
    }

    // Online softmax over kv (VGPRs + opposite half-wave).
    float bm = st[0][0];
    for (int r = 1; r < 8; r++) bm = fmaxf(bm, st[0][r]);
    for (int r = 0; r < 8; r++) bm = fmaxf(bm, st[1][r]);
    bm = fmaxf(bm, __shfl_xor(bm, 16, 32));
    const float mnew = fmaxf(mrow, bm);
    const float corr = __expf(mrow - mnew);
    float e[2][8];
    float psum = 0.f;
    for (int t = 0; t < 2; t++)
      for (int r = 0; r < 8; r++) {
        const float ev = __expf(st[t][r] - mnew);
        e[t][r] = ev;
        psum += ev;
      }
    psum += __shfl_xor(psum, 16, 32);
    lrow = lrow * corr + psum;
    mrow = mnew;
    for (int i = 0; i < 8; i++) o[i] *= corr;

    // P (D layout, 32x16) -> WMMA B fragment via pack + half-wave swap.
    unsigned pk0[4], pk1[4];
    for (int i = 0; i < 4; i++) {
      auto a0 = __builtin_amdgcn_cvt_pkrtz(e[0][2 * i], e[0][2 * i + 1]);
      auto a1 = __builtin_amdgcn_cvt_pkrtz(e[1][2 * i], e[1][2 * i + 1]);
      __builtin_memcpy(&pk0[i], &a0, 4);
      __builtin_memcpy(&pk1[i], &a1, 4);
    }
    union { v16h h; unsigned u[8]; } pf;
    for (int i = 0; i < 4; i++) {
      const unsigned s0 = __shfl_xor(pk0[i], 16, 32);
      const unsigned s1 = __shfl_xor(pk1[i], 16, 32);
      pf.u[i]     = (lane < 16) ? pk0[i] : s1;  // kv 0..7  | 16..23
      pf.u[i + 4] = (lane < 16) ? s0 : pk1[i];  // kv 8..15 | 24..31
    }

    // O^T += V^T_tile (A operand, contiguous) * P
    for (int dt = 0; dt < 8; dt++) {
      const _Float16* vp = vbh + (size_t)(dt * 16 + laneM) * SEQ + kv0 + laneKa;
      AV a;
      a.q[0] = *(const uint4*)vp;
      a.q[1] = *(const uint4*)(vp + 8);
      o[dt] = WMMA_F32_F16(a.h, pf.h, o[dt]);
    }
  }

  // Normalize and scatter real head dims 0..79 (f16 for the proj GEMM/TDM).
  const float inv = 1.f / lrow;
  const int b = bh >> 4, h = bh & 15;
  _Float16* obase = O + (size_t)(b * SEQ + q0 + laneM) * DIM + h * HD;
  const int rbase = laneKb >> 1;
  for (int dt = 0; dt < 5; dt++)
    for (int r = 0; r < 8; r++)
      obase[dt * 16 + rbase + r] = (_Float16)(o[dt][r] * inv);
}

// ---------------------------------------------------------------------------
extern "C" void kernel_launch(void* const* d_in, const int* in_sizes, int n_in,
                              void* d_out, int out_size, void* d_ws, size_t ws_size,
                              hipStream_t stream) {
  const float* x      = (const float*)d_in[0];
  // d_in[1] = cu_seqlens (unused by the reference computation)
  const float* rot    = (const float*)d_in[2];
  const float* qkv_w  = (const float*)d_in[3];
  const float* qkv_b  = (const float*)d_in[4];
  const float* proj_w = (const float*)d_in[5];
  const float* proj_b = (const float*)d_in[6];
  float* out = (float*)d_out;

  char* ws = (char*)d_ws;
  _Float16* Xh    = (_Float16*)(ws);                 // 8192*1280 f16 = 20,971,520 B
  _Float16* Oh    = Xh;                              // aliases Xh (dead after GEMM1)
  _Float16* Wqh   = (_Float16*)(ws + 20971520);      // 3840*1280 f16 =  9,830,400 B
  _Float16* Wph   = (_Float16*)(ws + 30801920);      // 1280*1280 f16 =  3,276,800 B
  _Float16* qkv16 = (_Float16*)(ws + 34078720);      // 8192*3840 f16 = 62,914,560 B
  _Float16* Qh    = (_Float16*)(ws + 96993280);      // 64*2048*128   = 33,554,432 B
  _Float16* Kh    = (_Float16*)(ws + 130547712);
  _Float16* Vt    = (_Float16*)(ws + 164102144);     // ends at 197,656,576 B

  // 0) f32 -> f16 for TDM-fed GEMM operands
  f32_to_f16<<<(NTOK * DIM) / 1024, 256, 0, stream>>>(x, Xh, NTOK * DIM);
  f32_to_f16<<<(3 * DIM * DIM) / 1024, 256, 0, stream>>>(qkv_w, Wqh, 3 * DIM * DIM);
  f32_to_f16<<<(DIM * DIM) / 1024, 256, 0, stream>>>(proj_w, Wph, DIM * DIM);

  // 1) qkv = x @ qkv_w^T + qkv_b  (f16 out)
  gemm_bias<<<dim3(3 * DIM / 128, NTOK / 128), 256, 0, stream>>>(
      Xh, Wqh, qkv_b, qkv16, nullptr, 3 * DIM, DIM);

  // 2) RoPE + pack into padded Q/K and transposed V
  rope_pack<<<(NTOK * NH * HDP) / 256, 256, 0, stream>>>(qkv16, rot, Qh, Kh, Vt);

  // 3) Flash attention per (batch, head, 16-row q block); f16 out into Oh
  attn<<<64 * (SEQ / 16), 32, 0, stream>>>(Qh, Kh, Vt, Oh);

  // 4) out = O @ proj_w^T + proj_b  (f32 out)
  gemm_bias<<<dim3(DIM / 128, NTOK / 128), 256, 0, stream>>>(
      Oh, Wph, proj_b, nullptr, out, DIM, DIM);
}
